// QuantumLSTMCell_53317724013116
// MI455X (gfx1250) — matrix operand order repaired and checked by reading, not compiled, not run on hardware
//
#include <hip/hip_runtime.h>
#include <math.h>

// ---------------------------------------------------------------------------
// QuantumLSTMCell for MI455X (gfx1250, wave32, WMMA)
//   Kernel A: pqc_in = tanh([x|h] @ W1^T + b1); z = 4-qubit/2-layer PQC sim
//   Kernel B: gates = [z | x|h] @ W2^T + b2  (bf16 WMMA, software-pipelined
//             256x(4x64) tiles, fused rank-4 z-update + bias + LSTM epilogue)
// ---------------------------------------------------------------------------

typedef __attribute__((ext_vector_type(16))) __bf16 v16bf;
typedef __attribute__((ext_vector_type(8)))  __bf16 v8bf;
typedef __attribute__((ext_vector_type(4)))  __bf16 v4bf;
typedef __attribute__((ext_vector_type(8)))  float  v8f;

#define DEVINL __device__ __attribute__((always_inline)) static inline

DEVINL float sigm(float v) { return 1.0f / (1.0f + __expf(-v)); }

// ----- tiny statevector helpers (16 complex amps, fully unrolled -> VGPRs) --
DEVINL void qgate(float* re, float* im, int w,
                  float ar, float ai, float br, float bi,
                  float cr, float ci, float dr, float di)
{
    const int st = 8 >> w;  // wire w acts on bit (3-w) of flat index
#pragma unroll
    for (int idx = 0; idx < 16; ++idx) {
        if (idx & st) continue;
        const int i1 = idx | st;
        const float s0r = re[idx], s0i = im[idx];
        const float s1r = re[i1],  s1i = im[i1];
        re[idx] = ar * s0r - ai * s0i + br * s1r - bi * s1i;
        im[idx] = ar * s0i + ai * s0r + br * s1i + bi * s1r;
        re[i1]  = cr * s0r - ci * s0i + dr * s1r - di * s1i;
        im[i1]  = cr * s0i + ci * s0r + dr * s1i + di * s1r;
    }
}

DEVINL void qcnot(float* re, float* im, int c, int t)
{
    const int cs = 8 >> c, ts = 8 >> t;
#pragma unroll
    for (int idx = 0; idx < 16; ++idx) {
        if (!(idx & cs) || (idx & ts)) continue;  // control=1, lower target idx
        const int i1 = idx | ts;
        float tr = re[idx]; re[idx] = re[i1]; re[i1] = tr;
        float ti = im[idx]; im[idx] = im[i1]; im[i1] = ti;
    }
}

// ---------------------------------------------------------------------------
// Kernel A: per-sample PQC. 64 samples/block, split-K(4) dot products.
// ---------------------------------------------------------------------------
__global__ __launch_bounds__(256) void pqc_kernel(
    const float* __restrict__ x, const float* __restrict__ hp,
    const float* __restrict__ W1, const float* __restrict__ b1,
    const float* __restrict__ qw, float* __restrict__ z)
{
    __shared__ float partial[64][4][4];  // [sample][kpart][q]
    __shared__ float th[64][4];          // tanh'd PQC input angles

    const int t    = threadIdx.x;
    const int sl   = t >> 2;             // local sample 0..63
    const int part = t & 3;              // K quarter 0..3
    const long s   = (long)blockIdx.x * 64 + sl;

    // pointer-select instead of data-select keeps loads as single b128 ops
    const float* px = x  + s * 512;
    const float* ph = hp + s * 1024 - 512;

    float acc[4] = {0.f, 0.f, 0.f, 0.f};
    const int kbeg = part * 384;
    for (int k = kbeg; k < kbeg + 384; k += 4) {
        const float* ap = (k < 512) ? (px + k) : (ph + k);
        const float4 a = *(const float4*)ap;
#pragma unroll
        for (int q = 0; q < 4; ++q) {
            const float4 w = *(const float4*)(W1 + q * 1536 + k);
            acc[q] += a.x * w.x + a.y * w.y + a.z * w.z + a.w * w.w;
        }
    }
#pragma unroll
    for (int q = 0; q < 4; ++q) partial[sl][part][q] = acc[q];
    __syncthreads();

    {   // thread `part` finalizes q = part
        const int q = part;
        const float v = partial[sl][0][q] + partial[sl][1][q] +
                        partial[sl][2][q] + partial[sl][3][q];
        th[sl][q] = tanhf(v + b1[q]);
    }
    __syncthreads();

    if (part == 0) {
        float re[16], im[16];
#pragma unroll
        for (int i = 0; i < 16; ++i) { re[i] = 0.f; im[i] = 0.f; }
        re[0] = 1.0f;

        // AngleEmbedding (RY)
#pragma unroll
        for (int w = 0; w < 4; ++w) {
            const float a = 0.5f * th[sl][w];
            const float c = cosf(a), sn = sinf(a);
            qgate(re, im, w, c, 0.f, -sn, 0.f, sn, 0.f, c, 0.f);
        }
#pragma unroll
        for (int l = 0; l < 2; ++l) {
            qcnot(re, im, 0, 1);
            qcnot(re, im, 1, 2);
            qcnot(re, im, 2, 3);
            qcnot(re, im, 3, 0);
#pragma unroll
            for (int i = 0; i < 4; ++i) {
                const float tx = 0.5f * qw[((l * 4 + i) * 3) + 0];
                const float ty = 0.5f * qw[((l * 4 + i) * 3) + 1];
                const float tz = 0.5f * qw[((l * 4 + i) * 3) + 2];
                float c, sn;
                c = cosf(tx); sn = sinf(tx);   // RX
                qgate(re, im, i, c, 0.f, 0.f, -sn, 0.f, -sn, c, 0.f);
                c = cosf(ty); sn = sinf(ty);   // RY
                qgate(re, im, i, c, 0.f, -sn, 0.f, sn, 0.f, c, 0.f);
                c = cosf(tz); sn = sinf(tz);   // RZ = diag(e^{-ia}, e^{+ia})
                qgate(re, im, i, c, -sn, 0.f, 0.f, 0.f, 0.f, c, sn);
            }
        }
#pragma unroll
        for (int q = 0; q < 4; ++q) {
            const int mask = 8 >> q;
            float zq = 0.f;
#pragma unroll
            for (int idx = 0; idx < 16; ++idx) {
                const float p = re[idx] * re[idx] + im[idx] * im[idx];
                zq += (idx & mask) ? -p : p;
            }
            z[s * 4 + q] = zq;
        }
    }
}

// ---------------------------------------------------------------------------
// Kernel B: bf16 WMMA GEMM (M=8192, N=4096 gate-interleaved, K=1536).
// Block tile: 256 M x (4 gates x 64 hidden cols). 512 thr = 16 waves,
// wave grid 4(M) x 4(N); wave tile 64x64 -> acc[4 m-tiles][4 gates].
// Register-staged prefetch into single-buffered LDS tiles; fully fused LSTM epilogue.
// ---------------------------------------------------------------------------
#define BM  256
#define HB  64   // hidden columns per block
#define BKB 32
#define LDK 48   // padded bf16 K-stride (96B rows, 16B-aligned fragments)

__global__ __launch_bounds__(512) void gemm_lstm_kernel(
    const float* __restrict__ x,  const float* __restrict__ hp,
    const float* __restrict__ cp, const float* __restrict__ W2,
    const float* __restrict__ b2, const float* __restrict__ z,
    float* __restrict__ hout, float* __restrict__ cout)
{
    __shared__ __align__(16) __bf16 As[BM * LDK];   // A[m][k] tile (bf16)
    __shared__ __align__(16) __bf16 Bs[BM * LDK];   // W2 rows, gate-interleaved

    const int tid   = threadIdx.x;
    const int lane  = tid & 31;
    const int wave  = tid >> 5;             // 0..15
    const int waveM = wave >> 2;            // 0..3 : 64 M rows each
    const int waveN = wave & 3;             // 0..3 : 16 hidden cols each
    const int hblk  = blockIdx.x * HB;      // hidden column block base
    const int mblk  = blockIdx.y * BM;

    v8f acc[4][4];
#pragma unroll
    for (int a = 0; a < 4; ++a)
#pragma unroll
        for (int g = 0; g < 4; ++g)
#pragma unroll
            for (int e = 0; e < 8; ++e) acc[a][g][e] = 0.f;

    // cooperative tile-load coordinates: 2 threads per tile row
    const int lr    = tid >> 1;             // 0..255
    const int lhalf = tid & 1;              // k-offset 0 / 16
    const int koff  = lhalf * 16;

    // B tile row -> global W2 row (gate-interleaved so the epilogue fuses):
    //   row = wn*64 + gate*16 + jcol  ->  W2 row gate*1024 + hblk + wn*16 + jcol
    const int gate  = (lr & 63) >> 4;       // 0..3
    const int wn    = lr >> 6;              // 0..3 (waveN ownership)
    const int jcol  = lr & 15;
    const float* w2base =
        W2 + (size_t)(gate * 1024 + hblk + wn * 16 + jcol) * 1540 + 4;
    const long am = (long)(mblk + lr);
    const float* px = x  + am * 512;            // A source, k in [0,512)
    const float* ph = hp + am * 1024 - 512;     // A source, k in [512,1536)

    // fragment coordinates (ISA 16-bit A/B layouts)
    const int khalf = lane >> 4;                // lane-half K group
    const int ml    = lane & 15;
    const int arow0 = (waveM * 64 + ml) * LDK;          // + tm*16*LDK
    const int brow0 = (waveN * 64 + ml) * LDK;          // + g*16*LDK

    float4 fa[4], fb[4];

    // ---- stage 0: first tile's global loads + LDS commit ----
#pragma unroll
    for (int v = 0; v < 4; ++v) {
        fa[v] = *(const float4*)(px + koff + v * 4);            // k0 = 0 -> x
        fb[v] = *(const float4*)(w2base + koff + v * 4);
    }
#pragma unroll
    for (int v = 0; v < 4; ++v) {
        v4bf pa, pb;
        pa[0] = (__bf16)fa[v].x; pa[1] = (__bf16)fa[v].y;
        pa[2] = (__bf16)fa[v].z; pa[3] = (__bf16)fa[v].w;
        pb[0] = (__bf16)fb[v].x; pb[1] = (__bf16)fb[v].y;
        pb[2] = (__bf16)fb[v].z; pb[3] = (__bf16)fb[v].w;
        *(v4bf*)&As[lr * LDK + koff + v * 4] = pa;
        *(v4bf*)&Bs[lr * LDK + koff + v * 4] = pb;
    }

    for (int k0 = 0; k0 < 1536; k0 += BKB) {
        __syncthreads();                    // LDS tile k0 ready

        // ---- prefetch tile k0+32 into registers (overlaps WMMA below) ----
        const bool more = (k0 + BKB) < 1536;
        if (more) {
            const int kn = k0 + BKB;
            const float* abase = (kn < 512) ? (px + kn) : (ph + kn);
#pragma unroll
            for (int v = 0; v < 4; ++v) {
                fa[v] = *(const float4*)(abase + koff + v * 4);
                fb[v] = *(const float4*)(w2base + kn + koff + v * 4);
            }
            __builtin_prefetch(w2base + kn + BKB, 0, 0);  // global_prefetch_b8
        }

        // ---- fragments + 16 WMMAs on tile k0 ----
        union { v16bf v; v8bf h[2]; } afrag[4], bfrag[4];
#pragma unroll
        for (int tm = 0; tm < 4; ++tm) {
            const int row = arow0 + tm * 16 * LDK;
            afrag[tm].h[0] = *(const v8bf*)&As[row + khalf * 8];
            afrag[tm].h[1] = *(const v8bf*)&As[row + 16 + khalf * 8];
        }
#pragma unroll
        for (int g = 0; g < 4; ++g) {
            const int row = brow0 + g * 16 * LDK;
            bfrag[g].h[0] = *(const v8bf*)&Bs[row + khalf * 8];
            bfrag[g].h[1] = *(const v8bf*)&Bs[row + 16 + khalf * 8];
        }
#pragma unroll
        for (int tm = 0; tm < 4; ++tm)
#pragma unroll
            for (int g = 0; g < 4; ++g)
                acc[tm][g] = __builtin_amdgcn_wmma_f32_16x16x32_bf16(
                    false, afrag[tm].v, false, bfrag[g].v,
                    (short)0, acc[tm][g], false, false);

        __syncthreads();                    // all waves done reading tile k0

        // ---- commit prefetched tile k0+32 to LDS ----
        if (more) {
#pragma unroll
            for (int v = 0; v < 4; ++v) {
                v4bf pa, pb;
                pa[0] = (__bf16)fa[v].x; pa[1] = (__bf16)fa[v].y;
                pa[2] = (__bf16)fa[v].z; pa[3] = (__bf16)fa[v].w;
                pb[0] = (__bf16)fb[v].x; pb[1] = (__bf16)fb[v].y;
                pb[2] = (__bf16)fb[v].z; pb[3] = (__bf16)fb[v].w;
                *(v4bf*)&As[lr * LDK + koff + v * 4] = pa;
                *(v4bf*)&Bs[lr * LDK + koff + v * 4] = pb;
            }
        }
    }

    // ---- fused epilogue: all 4 gates for (m, hcol) live in this lane ----
    const int hcol = hblk + waveN * 16 + ml;
#pragma unroll
    for (int tm = 0; tm < 4; ++tm) {
#pragma unroll
        for (int r = 0; r < 8; ++r) {
            const long m = (long)(mblk + waveM * 64 + tm * 16 +
                                  khalf * 8 + r);
            const float4 zv = *(const float4*)(z + m * 4);
            float gv[4];
#pragma unroll
            for (int g = 0; g < 4; ++g) {
                const size_t row = (size_t)(g * 1024 + hcol);
                const float4 wz = *(const float4*)(W2 + row * 1540);
                gv[g] = acc[tm][g][r] + zv.x * wz.x + zv.y * wz.y +
                        zv.z * wz.z + zv.w * wz.w + b2[row];
            }
            const float ft = sigm(gv[0]);
            const float it = sigm(gv[1]);
            const float ot = sigm(gv[2]);
            const float gt = tanhf(gv[3]);
            const float ct = ft * cp[m * 1024 + hcol] + it * gt;
            const float ht = ot * tanhf(ct);
            hout[m * 1024 + hcol] = ht;
            cout[m * 1024 + hcol] = ct;
        }
    }
}

// ---------------------------------------------------------------------------
extern "C" void kernel_launch(void* const* d_in, const int* in_sizes, int n_in,
                              void* d_out, int out_size, void* d_ws,
                              size_t ws_size, hipStream_t stream)
{
    const float* x  = (const float*)d_in[0];
    const float* hp = (const float*)d_in[1];
    const float* cp = (const float*)d_in[2];
    const float* W1 = (const float*)d_in[3];
    const float* b1 = (const float*)d_in[4];
    const float* qw = (const float*)d_in[5];
    const float* W2 = (const float*)d_in[6];
    const float* b2 = (const float*)d_in[7];
    (void)n_in; (void)out_size; (void)ws_size;

    const int B = in_sizes[0] / 512;            // 8192
    float* z    = (float*)d_ws;                 // B x 4 scratch
    float* hout = (float*)d_out;                // (h_t, c_t) concatenated
    float* cout = hout + (size_t)B * 1024;

    pqc_kernel<<<B / 64, 256, 0, stream>>>(x, hp, W1, b1, qw, z);

    dim3 grid(1024 / HB, B / BM);               // 16 hidden blocks x 32 M blocks
    gemm_lstm_kernel<<<grid, 512, 0, stream>>>(x, hp, cp, W2, b2, z,
                                               hout, cout);
}
